// RotaryAttention_18837726561069
// MI455X (gfx1250) — compile-verified, matrix operand order checked by reading, not run on hardware
//
#include <hip/hip_runtime.h>

// ---------------- problem constants ----------------
#define D_MODEL  1024
#define N_HEADS  16
#define HEAD_DIM 64
#define SEQ      2048
#define BATCH    2

typedef __attribute__((ext_vector_type(16))) __bf16 bf16x16;
typedef __attribute__((ext_vector_type(8)))  __bf16 bf16x8;
typedef __attribute__((ext_vector_type(8)))  float  f32x8;

// D = A(16x32 bf16) * B(32x16 bf16) + C(16x16 f32)
__device__ __forceinline__ f32x8 wmma_bf16(bf16x16 a, bf16x16 b, f32x8 c) {
  return __builtin_amdgcn_wmma_f32_16x16x32_bf16(false, a, false, b, (short)0, c,
                                                 false, false);
}

// K-index mapping shared by 16-bit A and B fragments (wave32):
//   lanes 0-15 : elems 0-7 -> K 0-7,  elems 8-15 -> K 16-23
//   lanes16-31 : elems 0-7 -> K 8-15, elems 8-15 -> K 24-31
// => each 8-elem half is K-contiguous starting at 8*sub and 16+8*sub.
// rowk points at (row_start + k_base); works for global bf16 and LDS.
__device__ __forceinline__ bf16x16 frag_bf16row(const __bf16* rowk, int sub) {
  bf16x8 a = *(const bf16x8*)(rowk + 8 * sub);
  bf16x8 b = *(const bf16x8*)(rowk + 16 + 8 * sub);
  bf16x16 f;
#pragma unroll
  for (int e = 0; e < 8; ++e) { f[e] = a[e]; f[e + 8] = b[e]; }
  return f;
}

// ============ kernel 0: one-shot fp32 -> bf16 conversion ============
__global__ void __launch_bounds__(256)
cvt_bf16_kernel(const float* __restrict__ src, __bf16* __restrict__ dst, int n) {
  int i = (blockIdx.x * 256 + threadIdx.x) * 8;
  if (i < n) {
    float4 a = *(const float4*)(src + i);
    float4 b = *(const float4*)(src + i + 4);
    bf16x8 o;
    o[0] = (__bf16)a.x; o[1] = (__bf16)a.y; o[2] = (__bf16)a.z; o[3] = (__bf16)a.w;
    o[4] = (__bf16)b.x; o[5] = (__bf16)b.y; o[6] = (__bf16)b.z; o[7] = (__bf16)b.w;
    *(bf16x8*)(dst + i) = o;
  }
}

// ---- epilogue for Q/K/V tiles: bias (+ RoPE for Q,K), store bf16 head-major
__device__ __forceinline__ void qkv_epilogue(f32x8 c, int n_global, int p, int b,
                                             int srow_base, float bvv,
                                             __bf16* __restrict__ dst) {
  const int h = n_global >> 6;
  const int d = n_global & 63;
  __bf16* base = dst + ((size_t)b * N_HEADS + h) * SEQ * HEAD_DIM + d;
  if (p < 2) {  // RoPE: pair (d, d^1) sits in adjacent lane
    const float fr = __powf(10000.0f, -(float)(d & ~1) / 64.0f);
#pragma unroll
    for (int r = 0; r < 8; ++r) {
      float v  = c[r] + bvv;
      float vp = __shfl_xor(v, 1, 32);       // partner channel (already biased)
      float ang = (float)(srow_base + r) * fr;
      float cv = __cosf(ang), sv = __sinf(ang);
      float o = (d & 1) ? (v * cv + vp * sv) : (v * cv - vp * sv);
      base[(size_t)(srow_base + r) * HEAD_DIM] = (__bf16)o;
    }
  } else {
#pragma unroll
    for (int r = 0; r < 8; ++r)
      base[(size_t)(srow_base + r) * HEAD_DIM] = (__bf16)(c[r] + bvv);
  }
}

// ============ kernel 1: fused QKV projection + RoPE (bf16 operands) ============
// grid: (SEQ/16, D_MODEL/128, BATCH*3), block 128; each wave owns a 16x32 tile
// (two independent accumulators sharing one A fragment -> hazard slots filled).
__global__ void __launch_bounds__(128)
qkv_rope_kernel(const __bf16* __restrict__ xb,
                const __bf16* __restrict__ wqb, const float* __restrict__ bq,
                const __bf16* __restrict__ wkb, const float* __restrict__ bk,
                const __bf16* __restrict__ wvb, const float* __restrict__ bv,
                __bf16* __restrict__ q_ws, __bf16* __restrict__ k_ws,
                __bf16* __restrict__ v_ws) {
  const int wave = threadIdx.x >> 5;
  const int lane = threadIdx.x & 31;
  const int m    = lane & 15;
  const int sub  = lane >> 4;

  const int stile  = blockIdx.x;
  const int ntile0 = (blockIdx.y * 4 + wave) * 2;   // 16-col units
  const int b = blockIdx.z / 3;
  const int p = blockIdx.z - 3 * b;                 // 0=Q 1=K 2=V

  const __bf16* W   = (p == 0) ? wqb : (p == 1) ? wkb : wvb;
  const float* bias = (p == 0) ? bq  : (p == 1) ? bk  : bv;
  __bf16* dst       = (p == 0) ? q_ws : (p == 1) ? k_ws : v_ws;

  const __bf16* xrow  = xb + ((size_t)b * SEQ + (size_t)stile * 16 + m) * D_MODEL;
  const __bf16* wrow0 = W + ((size_t)(ntile0 * 16) + m) * D_MODEL;
  const __bf16* wrow1 = wrow0 + (size_t)16 * D_MODEL;

  f32x8 c0 = {}, c1 = {};
#pragma unroll 4
  for (int k0 = 0; k0 < D_MODEL; k0 += 32) {
    bf16x16 af = frag_bf16row(xrow + k0, sub);
    c0 = wmma_bf16(af, frag_bf16row(wrow0 + k0, sub), c0);
    c1 = wmma_bf16(af, frag_bf16row(wrow1 + k0, sub), c1);
  }

  const int srow_base = stile * 16 + 8 * sub;
  const int n0 = ntile0 * 16 + m;
  qkv_epilogue(c0, n0,      p, b, srow_base, bias[n0],      dst);
  qkv_epilogue(c1, n0 + 16, p, b, srow_base, bias[n0 + 16], dst);
}

// ============ kernel 2: flash attention (online softmax) ============
// grid: (SEQ/64, BATCH*N_HEADS), block 128; wave w owns 16 query rows.
// S^T = K*Q^T so the S^T C-fragment (keys on rows, queries on lanes) maps
// element-for-element onto the A-fragment of P for P*V.  V is stored
// TRANSPOSED in LDS so V B-fragments are two contiguous 16B ds loads.
__global__ void __launch_bounds__(128)
attn_kernel(const __bf16* __restrict__ q_ws, const __bf16* __restrict__ k_ws,
            const __bf16* __restrict__ v_ws, __bf16* __restrict__ a_ws) {
  __shared__ __bf16 ldsK[32 * HEAD_DIM];        // row-major [32][64]
  __shared__ __bf16 ldsVt[HEAD_DIM * 32];       // transposed [64][32]

  const int wave = threadIdx.x >> 5;
  const int lane = threadIdx.x & 31;
  const int m    = lane & 15;
  const int sub  = lane >> 4;

  const int bh = blockIdx.y;                    // b*16 + h
  const int b  = bh >> 4;
  const int h  = bh & 15;
  const size_t head_off = (size_t)bh * SEQ * HEAD_DIM;
  const __bf16* kb = k_ws + head_off;
  const __bf16* vb = v_ws + head_off;
  const __bf16* qb = q_ws + head_off;

  const int qbase = blockIdx.x * 64 + wave * 16;

  // Q^T B-fragments (lane = query), loaded once: k-steps 0..31 and 32..63
  const __bf16* qrow = qb + (size_t)(qbase + m) * HEAD_DIM;
  const bf16x16 qf0 = frag_bf16row(qrow, sub);
  const bf16x16 qf1 = frag_bf16row(qrow + 32, sub);

  f32x8 o0 = {}, o1 = {}, o2 = {}, o3 = {};
  float mrun = -3.0e38f, lrun = 0.0f;

  for (int j = 0; j < SEQ; j += 32) {
    __syncthreads();                            // previous chunk consumed
    {                                           // cooperative K/V chunk -> LDS
      const unsigned* ksrc = (const unsigned*)(kb + (size_t)j * HEAD_DIM);
      unsigned* dK = (unsigned*)ldsK;
      for (int t = threadIdx.x; t < (32 * HEAD_DIM) / 2; t += 128)
        dK[t] = ksrc[t];
      const __bf16* vsrc = vb + (size_t)j * HEAD_DIM;
      for (int t = threadIdx.x; t < 256; t += 128) {   // V: 16B load, transpose
        int row = t >> 3;                       // 0..31 (key)
        int colg = (t & 7) * 8;                 // head-dim group
        bf16x8 v = *(const bf16x8*)(vsrc + row * HEAD_DIM + colg);
#pragma unroll
        for (int e = 0; e < 8; ++e) ldsVt[(colg + e) * 32 + row] = v[e];
      }
    }
    __syncthreads();
    if (j + 32 < SEQ) {                         // global_prefetch_b8 next chunk
      __builtin_prefetch(kb + (size_t)(j + 32) * HEAD_DIM + threadIdx.x * 16, 0, 0);
      __builtin_prefetch(vb + (size_t)(j + 32) * HEAD_DIM + threadIdx.x * 16, 0, 0);
    }

    // S^T tiles: s0 = keys j..j+15, s1 = keys j+16..j+31 (queries on lanes)
    f32x8 s0 = {}, s1 = {};
    {
      bf16x16 ka = frag_bf16row(ldsK + m * HEAD_DIM, sub);
      bf16x16 kc = frag_bf16row(ldsK + (16 + m) * HEAD_DIM, sub);
      s0 = wmma_bf16(ka, qf0, s0);
      s1 = wmma_bf16(kc, qf0, s1);
      ka = frag_bf16row(ldsK + m * HEAD_DIM + 32, sub);
      kc = frag_bf16row(ldsK + (16 + m) * HEAD_DIM + 32, sub);
      s0 = wmma_bf16(ka, qf1, s0);
      s1 = wmma_bf16(kc, qf1, s1);
    }

    // online softmax over this 32-key chunk (per-lane = per-query state)
    float mloc = -3.0e38f;
#pragma unroll
    for (int e = 0; e < 8; ++e) {
      s0[e] *= 0.125f;                          // 1/sqrt(64)
      s1[e] *= 0.125f;
      mloc = fmaxf(mloc, fmaxf(s0[e], s1[e]));
    }
    mloc = fmaxf(mloc, __shfl_xor(mloc, 16, 32));   // combine lane halves
    const float mnew = fmaxf(mrun, mloc);
    const float corr = __expf(mrun - mnew);
    float psum = 0.0f;
    bf16x16 pa;                                 // P in A-fragment layout directly
#pragma unroll
    for (int e = 0; e < 8; ++e) {
      float p0 = __expf(s0[e] - mnew);
      float p1 = __expf(s1[e] - mnew);
      psum += p0 + p1;
      pa[e]     = (__bf16)p0;
      pa[e + 8] = (__bf16)p1;
    }
    psum += __shfl_xor(psum, 16, 32);
    lrun = lrun * corr + psum;
    mrun = mnew;

    // rescale accumulators: broadcast corr of query (r + 8*sub) to all lanes
#pragma unroll
    for (int r = 0; r < 8; ++r) {
      float cr = __shfl(corr, r + 8 * sub, 32);
      o0[r] *= cr; o1[r] *= cr; o2[r] *= cr; o3[r] *= cr;
    }

    // out += P(16x32) * V(32x64); V B-frag = contiguous rows of ldsVt
    o0 = wmma_bf16(pa, frag_bf16row(ldsVt + (0 * 16 + m) * 32, sub), o0);
    o1 = wmma_bf16(pa, frag_bf16row(ldsVt + (1 * 16 + m) * 32, sub), o1);
    o2 = wmma_bf16(pa, frag_bf16row(ldsVt + (2 * 16 + m) * 32, sub), o2);
    o3 = wmma_bf16(pa, frag_bf16row(ldsVt + (3 * 16 + m) * 32, sub), o3);
  }

  // normalize + store bf16 into [B][S][D_MODEL] for the output projection
  const float inv = 1.0f / lrun;
  const int hcol = h * 64;
#pragma unroll
  for (int r = 0; r < 8; ++r) {
    float lv = __shfl(inv, r + 8 * sub, 32);
    int qg = qbase + r + 8 * sub;
    __bf16* orow = a_ws + ((size_t)b * SEQ + qg) * D_MODEL + hcol;
    orow[0 * 16 + m] = (__bf16)(o0[r] * lv);
    orow[1 * 16 + m] = (__bf16)(o1[r] * lv);
    orow[2 * 16 + m] = (__bf16)(o2[r] * lv);
    orow[3 * 16 + m] = (__bf16)(o3[r] * lv);
  }
}

// ============ kernel 3: output projection (bf16 x bf16 -> f32) ============
// grid: (BATCH*SEQ/16, D_MODEL/128), block 128; 16x32 tile per wave.
__global__ void __launch_bounds__(128)
oproj_kernel(const __bf16* __restrict__ a_ws, const __bf16* __restrict__ wob,
             const float* __restrict__ bo, float* __restrict__ out) {
  const int wave = threadIdx.x >> 5;
  const int lane = threadIdx.x & 31;
  const int m    = lane & 15;
  const int sub  = lane >> 4;

  const int rtile  = blockIdx.x;
  const int ntile0 = (blockIdx.y * 4 + wave) * 2;

  const __bf16* arow  = a_ws + ((size_t)rtile * 16 + m) * D_MODEL;
  const __bf16* wrow0 = wob + ((size_t)(ntile0 * 16) + m) * D_MODEL;
  const __bf16* wrow1 = wrow0 + (size_t)16 * D_MODEL;

  f32x8 c0 = {}, c1 = {};
#pragma unroll 4
  for (int k0 = 0; k0 < D_MODEL; k0 += 32) {
    bf16x16 af = frag_bf16row(arow + k0, sub);
    c0 = wmma_bf16(af, frag_bf16row(wrow0 + k0, sub), c0);
    c1 = wmma_bf16(af, frag_bf16row(wrow1 + k0, sub), c1);
  }

  const int n0 = ntile0 * 16 + m;
  const float b0 = bo[n0], b1 = bo[n0 + 16];
#pragma unroll
  for (int r = 0; r < 8; ++r) {
    size_t row = (size_t)rtile * 16 + r + 8 * sub;
    out[row * D_MODEL + n0]      = c0[r] + b0;
    out[row * D_MODEL + n0 + 16] = c1[r] + b1;
  }
}

// ============================ host launcher ============================
extern "C" void kernel_launch(void* const* d_in, const int* in_sizes, int n_in,
                              void* d_out, int out_size, void* d_ws, size_t ws_size,
                              hipStream_t stream) {
  const float* x  = (const float*)d_in[0];
  const float* wq = (const float*)d_in[1];
  const float* bq = (const float*)d_in[2];
  const float* wk = (const float*)d_in[3];
  const float* bk = (const float*)d_in[4];
  const float* wv = (const float*)d_in[5];
  const float* bv = (const float*)d_in[6];
  const float* wo = (const float*)d_in[7];
  const float* bo = (const float*)d_in[8];
  float* out = (float*)d_out;

  const size_t n  = (size_t)BATCH * SEQ * D_MODEL;   // 4 Mi
  const size_t wn = (size_t)D_MODEL * D_MODEL;       // 1 Mi
  __bf16* q_ws = (__bf16*)d_ws;                      // [B][H][S][64]
  __bf16* k_ws = q_ws + n;
  __bf16* v_ws = k_ws + n;
  __bf16* a_ws = v_ws + n;                           // [B][S][D] bf16
  __bf16* xb   = a_ws + n;                           // bf16 copies
  __bf16* wqb  = xb + n;
  __bf16* wkb  = wqb + wn;
  __bf16* wvb  = wkb + wn;
  __bf16* wob  = wvb + wn;

  // one-shot conversions (removes all v_cvt from GEMM hot loops)
  cvt_bf16_kernel<<<(int)(n  / (256 * 8)), 256, 0, stream>>>(x,  xb,  (int)n);
  cvt_bf16_kernel<<<(int)(wn / (256 * 8)), 256, 0, stream>>>(wq, wqb, (int)wn);
  cvt_bf16_kernel<<<(int)(wn / (256 * 8)), 256, 0, stream>>>(wk, wkb, (int)wn);
  cvt_bf16_kernel<<<(int)(wn / (256 * 8)), 256, 0, stream>>>(wv, wvb, (int)wn);
  cvt_bf16_kernel<<<(int)(wn / (256 * 8)), 256, 0, stream>>>(wo, wob, (int)wn);

  qkv_rope_kernel<<<dim3(SEQ / 16, D_MODEL / 128, BATCH * 3), 128, 0, stream>>>(
      xb, wqb, bq, wkb, bk, wvb, bv, q_ws, k_ws, v_ws);
  attn_kernel<<<dim3(SEQ / 64, BATCH * N_HEADS), 128, 0, stream>>>(
      q_ws, k_ws, v_ws, a_ws);
  oproj_kernel<<<dim3(BATCH * SEQ / 16, D_MODEL / 128), 128, 0, stream>>>(
      a_ws, wob, bo, out);
}
